// HMUStack_probing_49520972923460
// MI455X (gfx1250) — compile-verified, matrix operand order checked by reading, not run on hardware
//
#include <hip/hip_runtime.h>
#include <hip/hip_bf16.h>

// ---------------------------------------------------------------------------
// HMU stack: two quadratic layers + linear classifier.
// GEMMs via v_wmma_f32_16x16x32_bf16; B operands staged in LDS with
// global_load_async_to_lds_b128 (double-buffered, ASYNCcnt-synchronized).
// ---------------------------------------------------------------------------

typedef __bf16  v16bf __attribute__((ext_vector_type(16)));
typedef float   v8f   __attribute__((ext_vector_type(8)));

union Frag16 {                 // 16 bf16 = 32 bytes = two b128 loads
    v16bf v;
    uint4 u[2];
};

struct bh4 { __bf16 x, y, z, w; };   // packed 4 x bf16 (8B store)

#define BATCH   8192
#define D_IN    1024
#define N_H     2048
#define KRANK   4
#define N_CLS   1000
#define N_PAD   1008           // 63 * 16, zero-padded classifier weight rows
#define BN_EPS  1e-5f

// hmu GEMM tiling
#define WAVES    8
#define MBLK     256           // 8 waves * 32 rows
#define CHUNK_K  128           // K elems staged per LDS buffer
#define KSTEPS   4             // CHUNK_K / 32
#define NMAT     5             // mu + 4 v
#define LROW     136           // padded LDS row stride in elems (272B = 68dw, 68%64=4)
#define BUFE     (NMAT * 16 * LROW)   // elems per buffer

// BN reduction tiling
#define BN_RBLK  32            // row blocks (8192/256)
#define BN_RPB   256           // rows per block

// ---------------------------------------------------------------- utilities

__device__ __forceinline__ float wave_sum32(float v) {
    #pragma unroll
    for (int off = 16; off > 0; off >>= 1)
        v += __shfl_xor(v, off, 32);
    return v;
}

// f32 -> bf16 cast, 4 elems/thread, zero-fill tail (n_out must be %4)
__global__ void cast_bf16_kernel(const float* __restrict__ in, __bf16* __restrict__ out,
                                 long n_in, long n_out) {
    long i = ((long)blockIdx.x * blockDim.x + threadIdx.x) * 4;
    if (i >= n_out) return;
    float4 f;
    f.x = (i     < n_in) ? in[i]     : 0.f;
    f.y = (i + 1 < n_in) ? in[i + 1] : 0.f;
    f.z = (i + 2 < n_in) ? in[i + 2] : 0.f;
    f.w = (i + 3 < n_in) ? in[i + 3] : 0.f;
    bh4 b; b.x = (__bf16)f.x; b.y = (__bf16)f.y; b.z = (__bf16)f.z; b.w = (__bf16)f.w;
    *reinterpret_cast<bh4*>(out + i) = b;
}

// one wave per row: out[r] = sum_d X[r,d]^2
__global__ void rowsq_kernel(const float* __restrict__ X, int D, int rows,
                             float* __restrict__ out) {
    int wid  = ((int)blockIdx.x * blockDim.x + threadIdx.x) >> 5;
    int lane = threadIdx.x & 31;
    if (wid >= rows) return;
    const float* p = X + (long)wid * D;
    float s = 0.f;
    for (int d = lane; d < D; d += 32) { float x = p[d]; s = fmaf(x, x, s); }
    s = wave_sum32(s);
    if (lane == 0) out[wid] = s;
}

// one wave per n: coef[n] = { |mu_n|^2, mu_n.v_{n,0..3} }
__global__ void wcoef_kernel(const float* __restrict__ mu, const float* __restrict__ v,
                             int D, int N, float* __restrict__ coef) {
    int n    = ((int)blockIdx.x * blockDim.x + threadIdx.x) >> 5;
    int lane = threadIdx.x & 31;
    if (n >= N) return;
    const float* mp = mu + (long)n * D;
    const float* vp = v  + (long)n * KRANK * D;
    float sm = 0.f, s0 = 0.f, s1 = 0.f, s2 = 0.f, s3 = 0.f;
    for (int d = lane; d < D; d += 32) {
        float m = mp[d];
        sm = fmaf(m, m, sm);
        s0 = fmaf(m, vp[d],       s0);
        s1 = fmaf(m, vp[D + d],   s1);
        s2 = fmaf(m, vp[2*D + d], s2);
        s3 = fmaf(m, vp[3*D + d], s3);
    }
    sm = wave_sum32(sm); s0 = wave_sum32(s0); s1 = wave_sum32(s1);
    s2 = wave_sum32(s2); s3 = wave_sum32(s3);
    if (lane == 0) {
        float* c = coef + (long)n * 5;
        c[0] = sm; c[1] = s0; c[2] = s1; c[3] = s2; c[4] = s3;
    }
}

// BN stage 1: per (row-block, column) partial sum / sumsq. Fully coalesced.
__global__ void bn_partial_kernel(const float* __restrict__ in, int Nc,
                                  float* __restrict__ part) {   // (BN_RBLK, Nc, 2)
    int c  = (int)blockIdx.x * blockDim.x + threadIdx.x;
    int rb = (int)blockIdx.y;
    if (c >= Nc) return;
    const float* p = in + (long)rb * BN_RPB * Nc + c;
    float s = 0.f, ss = 0.f;
    #pragma unroll 4
    for (int r = 0; r < BN_RPB; ++r) {
        float x = p[(long)r * Nc];
        s += x; ss = fmaf(x, x, ss);
    }
    float* dst = part + ((long)rb * Nc + c) * 2;
    dst[0] = s; dst[1] = ss;
}

// BN stage 2: fold partials, emit affine stats = {a = rstd*gamma, c = beta - mean*a}
__global__ void bn_finish_kernel(const float* __restrict__ part, const float* __restrict__ gamma,
                                 const float* __restrict__ beta, int Nc, int rows,
                                 float* __restrict__ stats) {
    int c = (int)blockIdx.x * blockDim.x + threadIdx.x;
    if (c >= Nc) return;
    float s = 0.f, ss = 0.f;
    #pragma unroll
    for (int rb = 0; rb < BN_RBLK; ++rb) {
        const float* p = part + ((long)rb * Nc + c) * 2;
        s += p[0]; ss += p[1];
    }
    float inv = 1.0f / (float)rows;
    float m   = s * inv;
    float var = fmaxf(ss * inv - m * m, 0.f);   // biased variance (matches ref)
    float a   = rsqrtf(var + BN_EPS) * gamma[c];
    stats[2*c]   = a;
    stats[2*c+1] = beta[c] - m * a;
}

// y = a*x + c (opt relu); 4 columns/thread; f32 to 1-2 dests + packed bf16
template<int RELU>
__global__ void bn_apply_kernel(const float* __restrict__ in, const float* __restrict__ stats,
                                float* __restrict__ out1, float* __restrict__ out2,
                                __bf16* __restrict__ outb, int Nc, long total) {
    long i = ((long)blockIdx.x * blockDim.x + threadIdx.x) * 4;
    if (i >= total) return;
    int c = (int)(i % Nc);                     // Nc % 4 == 0 -> 4 consecutive columns
    float4 xv = *reinterpret_cast<const float4*>(in + i);
    float4 s0 = *reinterpret_cast<const float4*>(stats + 2 * c);      // a0,c0,a1,c1
    float4 s1 = *reinterpret_cast<const float4*>(stats + 2 * c + 4);  // a2,c2,a3,c3
    float4 y;
    y.x = fmaf(s0.x, xv.x, s0.y);
    y.y = fmaf(s0.z, xv.y, s0.w);
    y.z = fmaf(s1.x, xv.z, s1.y);
    y.w = fmaf(s1.z, xv.w, s1.w);
    if (RELU) {
        y.x = fmaxf(y.x, 0.f); y.y = fmaxf(y.y, 0.f);
        y.z = fmaxf(y.z, 0.f); y.w = fmaxf(y.w, 0.f);
    }
    *reinterpret_cast<float4*>(out1 + i) = y;
    if (out2) *reinterpret_cast<float4*>(out2 + i) = y;
    bh4 b; b.x = (__bf16)y.x; b.y = (__bf16)y.y; b.z = (__bf16)y.z; b.w = (__bf16)y.w;
    *reinterpret_cast<bh4*>(outb + i) = b;
}

// ------------------------------------------------- WMMA fragment addressing
// 16-bit A-layout (16x32 MxK), same pattern for B = W^T tiles (rows of W):
//   lanes 0-15 : row = lane,    halves at K+[0..7]  and K+[16..23]
//   lanes 16-31: row = lane-16, halves at K+[8..15] and K+[24..31]
__device__ __forceinline__ Frag16 load_frag(const __bf16* rowptr, int kt, int hi) {
    Frag16 f;
    const __bf16* p = rowptr + kt + hi * 8;
    f.u[0] = *reinterpret_cast<const uint4*>(p);
    f.u[1] = *reinterpret_cast<const uint4*>(p + 16);
    return f;
}

__device__ __forceinline__ v8f wmma_bf16(const Frag16& a, const Frag16& b, v8f c) {
    return __builtin_amdgcn_wmma_f32_16x16x32_bf16(false, a.v, false, b.v,
                                                   (short)0, c, false, false);
}

// ------------------------------------------------------------- HMU GEMM core
// Workgroup: 256 rows x 16 n; wave: 32 rows x 16 n (10 WMMAs / k-step).
// B tile (mu + 4 v, 16 rows each) double-buffered in LDS via async copy.
// MODE 0: out = -quad        MODE 1: out = exp(-quad/d)
template<int MODE>
__global__ __launch_bounds__(256)
void hmu_gemm_kernel(const __bf16* __restrict__ A,   // (BATCH, D) bf16
                     const __bf16* __restrict__ muW, // (N, D) bf16
                     const __bf16* __restrict__ vW,  // (N, K, D) bf16
                     const float* __restrict__ lam,
                     const float* __restrict__ coef, // (N,5): msq, c0..c3
                     const float* __restrict__ xsq,  // (BATCH,)
                     float* __restrict__ out,        // (BATCH, N) pre-BN
                     int D, int Nc, float inv_d) {
    __shared__ __align__(16) __bf16 smem[2 * BUFE];

    const int wave = threadIdx.x >> 5;
    const int lane = threadIdx.x & 31;
    const int nl   = lane & 15;
    const int hi   = lane >> 4;

    const int m0 = (int)blockIdx.y * MBLK + wave * 32;
    const int n0 = (int)blockIdx.x * 16;

    const __bf16* Ap0 = A + (long)(m0 + nl) * D;
    const __bf16* Ap1 = A + (long)(m0 + 16 + nl) * D;

    // --- per-wave async-copy slots: 10 rows of the 80-row B tile, 2 rows/issue
    unsigned long long grow[5];   // per-lane global byte address of row + k-part
    unsigned           lrow[5];   // per-lane LDS byte offset (buffer 0)
    const unsigned smbase = (unsigned)(uintptr_t)(&smem[0]);
    #pragma unroll
    for (int i = 0; i < 5; ++i) {
        int fr = wave * 10 + i * 2 + hi;                 // flat row 0..79
        int m  = fr >> 4;                                // matrix: 0=mu, 1..4=v_k
        int r  = fr & 15;                                // row within tile
        const __bf16* rb = (m == 0)
            ? (muW + (long)(n0 + r) * D)
            : (vW + (long)(n0 + r) * KRANK * D + (long)(m - 1) * D);
        grow[i] = (unsigned long long)(uintptr_t)rb + (unsigned)(nl * 16);
        lrow[i] = smbase + (unsigned)(fr * (LROW * 2) + nl * 16);
    }

    v8f accLo[NMAT] = {}, accHi[NMAT] = {};

    const int nch = D / CHUNK_K;

    // prefetch chunk 0 into buffer 0
    #pragma unroll
    for (int i = 0; i < 5; ++i) {
        asm volatile("global_load_async_to_lds_b128 %0, %1, off"
                     :: "v"(lrow[i]), "v"(grow[i]) : "memory");
    }

    for (int kc = 0; kc < nch; ++kc) {
        asm volatile("s_wait_asynccnt 0x0" ::: "memory");
        __syncthreads();                                  // chunk kc visible to all waves

        if (kc + 1 < nch) {                               // overlap: prefetch kc+1
            unsigned boff = ((kc + 1) & 1) ? (unsigned)(BUFE * 2) : 0u;
            unsigned long long kadd = (unsigned long long)(kc + 1) * (CHUNK_K * 2);
            #pragma unroll
            for (int i = 0; i < 5; ++i) {
                unsigned long long ga = grow[i] + kadd;
                unsigned la = lrow[i] + boff;
                asm volatile("global_load_async_to_lds_b128 %0, %1, off"
                             :: "v"(la), "v"(ga) : "memory");
            }
        }

        const __bf16* cur = smem + (kc & 1) * BUFE;
        const int kbase = kc * CHUNK_K;

        #pragma unroll
        for (int s = 0; s < KSTEPS; ++s) {
            int kt = kbase + s * 32;
            Frag16 a0 = load_frag(Ap0, kt, hi);
            Frag16 a1 = load_frag(Ap1, kt, hi);
            #pragma unroll
            for (int m = 0; m < NMAT; ++m) {
                const __bf16* p = cur + (m * 16 + nl) * LROW + s * 32 + hi * 8;
                Frag16 bm;
                bm.u[0] = *reinterpret_cast<const uint4*>(p);
                bm.u[1] = *reinterpret_cast<const uint4*>(p + 16);
                accLo[m] = wmma_bf16(a0, bm, accLo[m]);
                accHi[m] = wmma_bf16(a1, bm, accHi[m]);
            }
        }
    }

    // Epilogue: quad = lam*(xsq - 2g + msq) + sum_k (p_k - c_k)^2
    const int   n    = n0 + nl;
    const float lamn = lam[n];
    const float* cf  = coef + (long)n * 5;
    const float msqn = cf[0], c0 = cf[1], c1 = cf[2], c2 = cf[3], c3 = cf[4];

    #pragma unroll
    for (int g2i = 0; g2i < 2; ++g2i) {
        const v8f* acc = g2i ? accHi : accLo;
        int rbase = m0 + g2i * 16;
        #pragma unroll
        for (int r = 0; r < 8; ++r) {
            int   row = rbase + r + 8 * hi;   // C/D layout: VGPR r -> M=r (lo) / M=r+8 (hi)
            float g   = acc[0][r];
            float d0  = acc[1][r] - c0;
            float d1  = acc[2][r] - c1;
            float d2  = acc[3][r] - c2;
            float d3  = acc[4][r] - c3;
            float quad = lamn * (xsq[row] - 2.0f * g + msqn)
                       + d0*d0 + d1*d1 + d2*d2 + d3*d3;
            float val = (MODE == 0) ? -quad : __expf(-quad * inv_d);
            out[(long)row * Nc + n] = val;
        }
    }
}

// --------------------------------------------------------- classifier GEMM
__global__ __launch_bounds__(256)
void lin_gemm_kernel(const __bf16* __restrict__ A,  // (BATCH, N_H) bf16
                     const __bf16* __restrict__ W,  // (N_PAD, N_H) bf16, zero-padded
                     const float* __restrict__ bias,
                     float* __restrict__ out,       // (BATCH, N_CLS)
                     int D, int Nout) {
    const int wave = threadIdx.x >> 5;
    const int lane = threadIdx.x & 31;
    const int nl   = lane & 15;
    const int hi   = lane >> 4;

    const int m0 = (int)blockIdx.y * MBLK + wave * 32;
    const int n0 = (int)blockIdx.x * 16;

    const __bf16* Ap0 = A + (long)(m0 + nl) * D;
    const __bf16* Ap1 = A + (long)(m0 + 16 + nl) * D;
    const __bf16* Bp  = W + (long)(n0 + nl) * D;

    v8f acc0 = {}, acc1 = {};
    for (int kt = 0; kt < D; kt += 32) {
        Frag16 a0 = load_frag(Ap0, kt, hi);
        Frag16 a1 = load_frag(Ap1, kt, hi);
        Frag16 b  = load_frag(Bp,  kt, hi);
        acc0 = wmma_bf16(a0, b, acc0);
        acc1 = wmma_bf16(a1, b, acc1);
    }

    const int n = n0 + nl;
    if (n < Nout) {
        float bn = bias[n];
        #pragma unroll
        for (int r = 0; r < 8; ++r) {
            int row = m0 + r + 8 * hi;
            out[(long)row * Nout + n] = acc0[r] + bn;
            out[(long)(row + 16) * Nout + n] = acc1[r] + bn;
        }
    }
}

// ---------------------------------------------------------------------------
extern "C" void kernel_launch(void* const* d_in, const int* in_sizes, int n_in,
                              void* d_out, int out_size, void* d_ws, size_t ws_size,
                              hipStream_t stream) {
    (void)in_sizes; (void)n_in; (void)out_size; (void)ws_size;

    const float* x    = (const float*)d_in[0];
    const float* mu1  = (const float*)d_in[1];
    const float* lam1 = (const float*)d_in[2];
    const float* v1   = (const float*)d_in[3];
    const float* g1   = (const float*)d_in[4];
    const float* b1   = (const float*)d_in[5];
    const float* mu2  = (const float*)d_in[6];
    const float* lam2 = (const float*)d_in[7];
    const float* v2   = (const float*)d_in[8];
    const float* g2   = (const float*)d_in[9];
    const float* b2   = (const float*)d_in[10];
    const float* Wl   = (const float*)d_in[11];
    const float* bl   = (const float*)d_in[12];

    // ---- d_out layout: logits | feat1 | feat2 | feat2 (flat, f32)
    float* o       = (float*)d_out;
    float* logits  = o;
    float* feat1   = o + (long)BATCH * N_CLS;
    float* feat2a  = feat1  + (long)BATCH * N_H;
    float* feat2b  = feat2a + (long)BATCH * N_H;

    // ---- workspace carve-up
    char* ws = (char*)d_ws;
    size_t off = 0;
    auto carve = [&](size_t bytes) { void* p = ws + off; off = (off + bytes + 255) & ~(size_t)255; return p; };
    __bf16* xb   = (__bf16*)carve((size_t)BATCH * D_IN  * 2);
    __bf16* mu1b = (__bf16*)carve((size_t)N_H   * D_IN  * 2);
    __bf16* v1b  = (__bf16*)carve((size_t)N_H * KRANK * D_IN * 2);
    __bf16* f1b  = (__bf16*)carve((size_t)BATCH * N_H   * 2);
    __bf16* mu2b = (__bf16*)carve((size_t)N_H   * N_H   * 2);
    __bf16* v2b  = (__bf16*)carve((size_t)N_H * KRANK * N_H * 2);
    __bf16* f2b  = (__bf16*)carve((size_t)BATCH * N_H   * 2);
    __bf16* wlb  = (__bf16*)carve((size_t)N_PAD * N_H   * 2);
    float*  xsq1 = (float*)carve((size_t)BATCH * 4);
    float*  xsq2 = (float*)carve((size_t)BATCH * 4);
    float*  coef1= (float*)carve((size_t)N_H * 5 * 4);
    float*  coef2= (float*)carve((size_t)N_H * 5 * 4);
    float*  stats= (float*)carve((size_t)N_H * 2 * 4);
    float*  part = (float*)carve((size_t)BN_RBLK * N_H * 2 * 4);

    const int T = 256;
    auto blocks = [](long n, int t) { return (unsigned)((n + t - 1) / t); };

    // 1) bf16 casts (vectorized x4)
    { long n = (long)BATCH * D_IN;
      cast_bf16_kernel<<<blocks(n / 4, T), T, 0, stream>>>(x, xb, n, n); }
    { long n = (long)N_H * D_IN;
      cast_bf16_kernel<<<blocks(n / 4, T), T, 0, stream>>>(mu1, mu1b, n, n); }
    { long n = (long)N_H * KRANK * D_IN;
      cast_bf16_kernel<<<blocks(n / 4, T), T, 0, stream>>>(v1, v1b, n, n); }
    { long n = (long)N_H * N_H;
      cast_bf16_kernel<<<blocks(n / 4, T), T, 0, stream>>>(mu2, mu2b, n, n); }
    { long n = (long)N_H * KRANK * N_H;
      cast_bf16_kernel<<<blocks(n / 4, T), T, 0, stream>>>(v2, v2b, n, n); }
    { long nin = (long)N_CLS * N_H, nout = (long)N_PAD * N_H;
      cast_bf16_kernel<<<blocks(nout / 4, T), T, 0, stream>>>(Wl, wlb, nin, nout); }

    // 2) layer-1 constants
    rowsq_kernel<<<blocks((long)BATCH * 32, T), T, 0, stream>>>(x, D_IN, BATCH, xsq1);
    wcoef_kernel<<<blocks((long)N_H * 32, T), T, 0, stream>>>(mu1, v1, D_IN, N_H, coef1);

    // 3) layer-1 fused 5-way WMMA GEMM -> pre-BN into feat1 slab
    {
        dim3 grid(N_H / 16, BATCH / MBLK);
        hmu_gemm_kernel<0><<<grid, T, 0, stream>>>(xb, mu1b, v1b, lam1, coef1, xsq1,
                                                   feat1, D_IN, N_H, 0.f);
    }

    // 4) BatchNorm 1 (two-stage stats + vectorized apply)
    {
        dim3 g1d(N_H / T, BN_RBLK);
        bn_partial_kernel<<<g1d, T, 0, stream>>>(feat1, N_H, part);
        bn_finish_kernel<<<blocks(N_H, T), T, 0, stream>>>(part, g1, b1, N_H, BATCH, stats);
        bn_apply_kernel<0><<<blocks((long)BATCH * N_H / 4, T), T, 0, stream>>>(
            feat1, stats, feat1, nullptr, f1b, N_H, (long)BATCH * N_H);
    }

    // 5) layer-2 constants + fused GEMM (exp epilogue)
    rowsq_kernel<<<blocks((long)BATCH * 32, T), T, 0, stream>>>(feat1, N_H, BATCH, xsq2);
    wcoef_kernel<<<blocks((long)N_H * 32, T), T, 0, stream>>>(mu2, v2, N_H, N_H, coef2);
    {
        dim3 grid(N_H / 16, BATCH / MBLK);
        hmu_gemm_kernel<1><<<grid, T, 0, stream>>>(f1b, mu2b, v2b, lam2, coef2, xsq2,
                                                   feat2a, N_H, N_H, 1.0f / (float)N_H);
    }

    // 6) BatchNorm 2 + ReLU
    {
        dim3 g1d(N_H / T, BN_RBLK);
        bn_partial_kernel<<<g1d, T, 0, stream>>>(feat2a, N_H, part);
        bn_finish_kernel<<<blocks(N_H, T), T, 0, stream>>>(part, g2, b2, N_H, BATCH, stats);
        bn_apply_kernel<1><<<blocks((long)BATCH * N_H / 4, T), T, 0, stream>>>(
            feat2a, stats, feat2a, feat2b, f2b, N_H, (long)BATCH * N_H);
    }

    // 7) classifier GEMM
    {
        dim3 grid(N_PAD / 16, BATCH / MBLK);
        lin_gemm_kernel<<<grid, T, 0, stream>>>(f2b, wlb, bl, logits, N_H, N_CLS);
    }
}
